// FCOS_70068096467655
// MI455X (gfx1250) — matrix operand order repaired and checked by reading, not compiled, not run on hardware
//
#include <hip/hip_runtime.h>
#include <hip/hip_bf16.h>

typedef __attribute__((ext_vector_type(16))) __bf16        v16bf;
typedef __attribute__((ext_vector_type(8)))  float         v8f;
typedef __attribute__((ext_vector_type(8)))  unsigned int  v8u;

__device__ __forceinline__ unsigned short f2bf(float f) {
    unsigned int u = __float_as_uint(f);
    unsigned int r = u + 0x7FFFu + ((u >> 16) & 1u);
    return (unsigned short)(r >> 16);
}
__device__ __forceinline__ float bf2f(unsigned short b) {
    return __uint_as_float(((unsigned int)b) << 16);
}

// ---------------------------------------------------------------------------
// Pack conv weights  w[L][256][256][3][3] (f32, OIHW)  ->
//   out[L][9][128][256] (u32, each u32 = bf16 pair {cin=2cp, cin=2cp+1} for cout)
// Each WMMA B-matrix VGPR becomes a single dword load.
// ---------------------------------------------------------------------------
__global__ void fcos_pack_w(const float* __restrict__ w, unsigned int* __restrict__ out,
                            int layers) {
    int t = blockIdx.x * blockDim.x + threadIdx.x;
    int total = layers * 9 * 128 * 256;
    if (t >= total) return;
    int co   = t & 255;
    int rest = t >> 8;
    int cp   = rest & 127;
    rest   >>= 7;
    int tap  = rest % 9;
    int l    = rest / 9;
    const float* wl = w + (size_t)l * 256 * 256 * 9;
    float f0 = wl[((size_t)co * 256 + 2 * cp + 0) * 9 + tap];
    float f1 = wl[((size_t)co * 256 + 2 * cp + 1) * 9 + tap];
    out[t] = (unsigned int)f2bf(f0) | ((unsigned int)f2bf(f1) << 16);
}

// ---------------------------------------------------------------------------
// f32 NCHW feature -> bf16 NHWC, duplicated into cls-tower and box-tower bufs
// ---------------------------------------------------------------------------
__global__ void fcos_cvt_feat(const float* __restrict__ x,
                              unsigned short* __restrict__ a,
                              unsigned short* __restrict__ b, int HW) {
    int t = blockIdx.x * blockDim.x + threadIdx.x;
    if (t >= 2 * HW * 256) return;
    int c    = t & 255;
    int rest = t >> 8;            // n*HW + pix
    int pix  = rest % HW;
    int n    = rest / HW;
    unsigned short v = f2bf(x[((size_t)(n * 256 + c)) * HW + pix]);
    a[t] = v;
    b[t] = v;
}

// ---------------------------------------------------------------------------
// Implicit-GEMM 3x3 conv, 256 -> 256, SAME padding, via WMMA bf16.
//   act : bf16 NHWC [totalPx][256]      (totalPx = 2*H*W, batch-major)
//   wpk : u32 [9][128][256] packed bf16 pairs (see fcos_pack_w)
//   out : f32 [totalPx][256] = conv + bias
//
// Block: 256 threads = 8 waves. Block tile: 64 pixels (4 M-subtiles) x 256
// couts. Wave wv owns couts [32wv, 32wv+32): per K-chunk it loads its two B
// fragments once and replays them against 4 A fragments from LDS
// -> 8 WMMAs per 16 B dword loads. The cooperative A-tile load for the NEXT
// K-chunk is prefetched into registers right after the publishing barrier, so
// its VMEM latency overlaps the current chunk's WMMA burst.
// ---------------------------------------------------------------------------
__global__ __launch_bounds__(256) void fcos_conv_wmma(
        const unsigned short* __restrict__ act,
        const unsigned int*  __restrict__ wpk,
        const float*         __restrict__ bias,
        float*               __restrict__ out,
        int H, int W, int totalPx) {
    __shared__ unsigned int ldsA[64 * 16];   // 64 px x 32 bf16 (as 16 u32/px)

    const int tid  = threadIdx.x;
    const int lane = tid & 31;
    const int wv   = tid >> 5;               // wave 0..7
    const int m    = lane & 15;              // M row within 16x16 tile
    const int hiH  = lane >> 4;              // lane half selects K sub-offset
    const int HW   = H * W;
    const int blockPx = blockIdx.x * 64;

    // A-tile cooperative load mapping: thread t -> 4 pixels {p0+16j}, pair cpA
    const int p0  = tid >> 4;                // base pixel 0..15
    const int cpA = tid & 15;                // u32 (= 2 channels) within chunk

    int  hA[4], wA[4], nbase[4];
    bool okA[4];
#pragma unroll
    for (int j = 0; j < 4; ++j) {
        int g = blockPx + p0 + 16 * j;
        okA[j] = g < totalPx;
        int gg  = okA[j] ? g : 0;
        int n   = gg / HW;
        int pix = gg - n * HW;
        hA[j] = pix / W;
        wA[j] = pix - hA[j] * W;
        nbase[j] = n * HW;
    }

    v8f acc[4][2];
#pragma unroll
    for (int j = 0; j < 4; ++j) { acc[j][0] = (v8f){}; acc[j][1] = (v8f){}; }

    const int co0 = wv * 32 + m;             // first 16x16 B tile column
    const int kh2 = hiH * 4;                 // B: lane half -> K+8 (4 u32 rows)
    const int aSub = m * 16 + hiH * 4;       // A frag base within an M-subtile

    for (int tap = 0; tap < 9; ++tap) {
        const int dy = tap / 3 - 1, dx = tap % 3 - 1;
        const unsigned int* srcPx[4];
        bool inb[4];
#pragma unroll
        for (int j = 0; j < 4; ++j) {
            int yy = hA[j] + dy, xx = wA[j] + dx;
            inb[j] = okA[j] && yy >= 0 && yy < H && xx >= 0 && xx < W;
            srcPx[j] = (const unsigned int*)act +
                       (size_t)(nbase[j] + yy * W + xx) * 128;
        }

        // prefetch first chunk of this tap
        unsigned int av[4];
#pragma unroll
        for (int j = 0; j < 4; ++j)
            av[j] = inb[j] ? srcPx[j][cpA] : 0u;

#pragma unroll
        for (int ci = 0; ci < 8; ++ci) {      // cb = 32*ci
            __syncthreads();                  // everyone done reading old tile
#pragma unroll
            for (int j = 0; j < 4; ++j)
                ldsA[tid + 256 * j] = av[j];
            __syncthreads();                  // tile published

            // prefetch NEXT chunk: latency hides behind this chunk's WMMAs
            if (ci < 7) {
#pragma unroll
                for (int j = 0; j < 4; ++j)
                    av[j] = inb[j] ? srcPx[j][(ci + 1) * 16 + cpA] : 0u;
            }

            // B fragments for this wave: one dword load per VGPR
            const unsigned int* wb = wpk + ((size_t)tap * 128 + ci * 16 + kh2) * 256;
            v8u bu0, bu1;
#pragma unroll
            for (int v = 0; v < 8; ++v) {
                bu0[v] = wb[(v + (v & 4)) * 256 + co0];
                bu1[v] = wb[(v + (v & 4)) * 256 + co0 + 16];
            }
            v16bf b0 = __builtin_bit_cast(v16bf, bu0);
            v16bf b1 = __builtin_bit_cast(v16bf, bu1);

            // 4 A fragments from LDS, 2 WMMAs each
#pragma unroll
            for (int j = 0; j < 4; ++j) {
                v8u au;
#pragma unroll
                for (int v = 0; v < 4; ++v) {
                    au[v]     = ldsA[256 * j + aSub + v];
                    au[4 + v] = ldsA[256 * j + aSub + 8 + v];
                }
                v16bf a = __builtin_bit_cast(v16bf, au);
                acc[j][0] = __builtin_amdgcn_wmma_f32_16x16x32_bf16(
                    false, a, false, b0, (short)0, acc[j][0], false, false);
                acc[j][1] = __builtin_amdgcn_wmma_f32_16x16x32_bf16(
                    false, a, false, b1, (short)0, acc[j][1], false, false);
            }
        }
    }

    float b0 = bias[co0];
    float b1 = bias[co0 + 16];
#pragma unroll
    for (int j = 0; j < 4; ++j) {
#pragma unroll
        for (int r = 0; r < 8; ++r) {
            int row = hiH ? (8 + r) : r;     // ISA C/D layout
            int g   = blockPx + 16 * j + row;
            if (g < totalPx) {
                out[(size_t)g * 256 + co0]      = acc[j][0][r] + b0;
                out[(size_t)g * 256 + co0 + 16] = acc[j][1][r] + b1;
            }
        }
    }
}

// ---------------------------------------------------------------------------
// GroupNorm stats: 64 blocks = (n in 0..1) x (group in 0..31); 8 ch/group
// conv layout: f32 [2*HW][256]
// ---------------------------------------------------------------------------
__global__ __launch_bounds__(256) void fcos_gn_stats(const float* __restrict__ conv,
                                                     float* __restrict__ mean,
                                                     float* __restrict__ rstd, int HW) {
    int n = blockIdx.x >> 5;
    int g = blockIdx.x & 31;
    int count = HW * 8;
    float s = 0.f, s2 = 0.f;
    for (int i = threadIdx.x; i < count; i += 256) {
        int pix = i >> 3;
        int c   = (g << 3) + (i & 7);
        float v = conv[((size_t)(n * HW + pix)) * 256 + c];
        s  += v;
        s2 += v * v;
    }
    __shared__ float sh[256], sh2[256];
    sh[threadIdx.x]  = s;
    sh2[threadIdx.x] = s2;
    __syncthreads();
    for (int st = 128; st > 0; st >>= 1) {
        if (threadIdx.x < st) {
            sh[threadIdx.x]  += sh[threadIdx.x + st];
            sh2[threadIdx.x] += sh2[threadIdx.x + st];
        }
        __syncthreads();
    }
    if (threadIdx.x == 0) {
        float mu  = sh[0] / (float)count;
        float var = sh2[0] / (float)count - mu * mu;
        mean[blockIdx.x] = mu;
        rstd[blockIdx.x] = rsqrtf(var + 1e-5f);
    }
}

// GN apply + ReLU + bf16 NHWC repack (feeds next conv)
__global__ void fcos_gn_apply(const float* __restrict__ conv,
                              const float* __restrict__ mean,
                              const float* __restrict__ rstd,
                              const float* __restrict__ gw,
                              const float* __restrict__ gb,
                              unsigned short* __restrict__ actout,
                              int HW, int total) {
    int t = blockIdx.x * blockDim.x + threadIdx.x;
    if (t >= total) return;
    int c   = t & 255;
    int idx = t >> 8;                 // n*HW + pix
    int n   = idx / HW;
    int g   = n * 32 + (c >> 3);
    float y = (conv[t] - mean[g]) * rstd[g] * gw[c] + gb[c];
    actout[t] = f2bf(fmaxf(y, 0.f));
}

// ---------------------------------------------------------------------------
// Final head convs: score(80, from cls act), pred(4) + ctr(1, from box act)
// out85: f32 [totalPx][85]  (0..79 score, 80..83 pred, 84 ctr)
// ---------------------------------------------------------------------------
__global__ void fcos_head_conv(const unsigned short* __restrict__ actCls,
                               const unsigned short* __restrict__ actBox,
                               const float* __restrict__ score_w,
                               const float* __restrict__ score_b,
                               const float* __restrict__ pred_w,
                               const float* __restrict__ pred_b,
                               const float* __restrict__ ctr_w,
                               const float* __restrict__ ctr_b,
                               float* __restrict__ out85, int H, int W) {
    int t = blockIdx.x * blockDim.x + threadIdx.x;
    int HW = H * W;
    int totalPx = 2 * HW;
    if (t >= totalPx * 85) return;
    int co  = t % 85;
    int g   = t / 85;
    int n   = g / HW;
    int pix = g % HW;
    int h = pix / W, w = pix % W;

    const unsigned short* act;
    const float* wp;
    float acc;
    if (co < 80)      { act = actCls; wp = score_w + (size_t)co * 2304;        acc = score_b[co];      }
    else if (co < 84) { act = actBox; wp = pred_w  + (size_t)(co - 80) * 2304; acc = pred_b[co - 80];  }
    else              { act = actBox; wp = ctr_w;                              acc = ctr_b[0];         }

    for (int tap = 0; tap < 9; ++tap) {
        int yy = h + tap / 3 - 1, xx = w + tap % 3 - 1;
        if (yy < 0 || yy >= H || xx < 0 || xx >= W) continue;
        const unsigned short* ap = act + (size_t)(n * HW + yy * W + xx) * 256;
        for (int ci = 0; ci < 256; ++ci)
            acc += bf2f(ap[ci]) * wp[ci * 9 + tap];
    }
    out85[(size_t)g * 85 + co] = acc;
}

// Assemble: sigmoid(score)*sigmoid(ctr) + box decode -> d_out [2][21330][84]
__global__ void fcos_assemble(const float* __restrict__ out85,
                              float* __restrict__ dst,
                              const float* __restrict__ scales,
                              int level, int H, int W, int stride,
                              int rowBase, int totalRows) {
    int t = blockIdx.x * blockDim.x + threadIdx.x;
    int HW = H * W;
    int totalPx = 2 * HW;
    if (t >= totalPx * 84) return;
    int k   = t % 84;
    int g   = t / 84;
    int n   = g / HW;
    int pix = g % HW;
    int h = pix / W, w = pix % W;
    const float* src = out85 + (size_t)g * 85;

    float outv;
    if (k < 80) {
        float sc = 1.f / (1.f + __expf(-src[k]));
        float ct = 1.f / (1.f + __expf(-src[84]));
        outv = sc * ct;
    } else {
        int j = k - 80;
        float r  = fmaxf(src[80 + j] * scales[level], 0.f) * (float)stride;
        float sx = (float)w * (float)stride;
        float sy = (float)h * (float)stride;
        outv = (j == 0) ? sx - r : (j == 1) ? sy - r : (j == 2) ? sx + r : sy + r;
    }
    dst[((size_t)n * totalRows + rowBase + pix) * 84 + k] = outv;
}

// ---------------------------------------------------------------------------
extern "C" void kernel_launch(void* const* d_in, const int* in_sizes, int n_in,
                              void* d_out, int out_size, void* d_ws, size_t ws_size,
                              hipStream_t stream) {
    (void)in_sizes; (void)n_in; (void)out_size; (void)ws_size;

    const float* feats[5];
    for (int l = 0; l < 5; ++l) feats[l] = (const float*)d_in[l];
    const float* cls_w   = (const float*)d_in[5];
    const float* cls_b   = (const float*)d_in[6];
    const float* cls_gw  = (const float*)d_in[7];
    const float* cls_gb  = (const float*)d_in[8];
    const float* box_w   = (const float*)d_in[9];
    const float* box_b   = (const float*)d_in[10];
    const float* box_gw  = (const float*)d_in[11];
    const float* box_gb  = (const float*)d_in[12];
    const float* score_w = (const float*)d_in[13];
    const float* score_b = (const float*)d_in[14];
    const float* pred_w  = (const float*)d_in[15];
    const float* pred_b  = (const float*)d_in[16];
    const float* ctr_w   = (const float*)d_in[17];
    const float* ctr_b   = (const float*)d_in[18];
    const float* scales  = (const float*)d_in[19];

    // ---- workspace carve-up (sizes are multiples of 256 B) ----
    char* ws = (char*)d_ws;
    size_t off = 0;
    const size_t WPK_U32_PER_LAYER = (size_t)9 * 128 * 256;           // 294,912
    unsigned int* wclsPk = (unsigned int*)(ws + off); off += 4 * WPK_U32_PER_LAYER * 4;
    unsigned int* wboxPk = (unsigned int*)(ws + off); off += 4 * WPK_U32_PER_LAYER * 4;
    const size_t ACT_BYTES = (size_t)2 * 100 * 160 * 256 * 2;          // 16.38 MB
    unsigned short* actA = (unsigned short*)(ws + off); off += ACT_BYTES;
    unsigned short* actB = (unsigned short*)(ws + off); off += ACT_BYTES;
    unsigned short* actC = (unsigned short*)(ws + off); off += ACT_BYTES;
    unsigned short* actD = (unsigned short*)(ws + off); off += ACT_BYTES;
    float* convf = (float*)(ws + off); off += (size_t)2 * 100 * 160 * 256 * 4;  // 32.77 MB
    float* gmean = (float*)(ws + off); off += 256;
    float* grstd = (float*)(ws + off); off += 256;

    // ---- pack tower weights to bf16 WMMA-friendly layout ----
    {
        int total = 4 * 9 * 128 * 256;
        int blocks = (total + 255) / 256;
        fcos_pack_w<<<blocks, 256, 0, stream>>>(cls_w, wclsPk, 4);
        fcos_pack_w<<<blocks, 256, 0, stream>>>(box_w, wboxPk, 4);
    }

    const int Hs[5] = {100, 50, 25, 13, 7};
    const int Ws2[5] = {160, 80, 40, 20, 10};
    const int STRD[5] = {8, 16, 32, 64, 128};
    const int totalRows = 21330;

    int rowBase = 0;
    for (int l = 0; l < 5; ++l) {
        const int H = Hs[l], W = Ws2[l], HW = H * W, totalPx = 2 * HW;

        fcos_cvt_feat<<<(totalPx * 256 + 255) / 256, 256, 0, stream>>>(
            feats[l], actA, actC, HW);

        unsigned short *cin = actA, *cout_ = actB;
        unsigned short *bin = actC, *bout_ = actD;
        const int convBlocks = (totalPx + 63) / 64;
        const int elemBlocks = (totalPx * 256 + 255) / 256;

        for (int i = 0; i < 4; ++i) {
            // cls tower layer
            fcos_conv_wmma<<<convBlocks, 256, 0, stream>>>(
                cin, wclsPk + (size_t)i * WPK_U32_PER_LAYER, cls_b + i * 256,
                convf, H, W, totalPx);
            fcos_gn_stats<<<64, 256, 0, stream>>>(convf, gmean, grstd, HW);
            fcos_gn_apply<<<elemBlocks, 256, 0, stream>>>(
                convf, gmean, grstd, cls_gw + i * 256, cls_gb + i * 256,
                cout_, HW, totalPx * 256);
            { unsigned short* t = cin; cin = cout_; cout_ = t; }

            // box tower layer
            fcos_conv_wmma<<<convBlocks, 256, 0, stream>>>(
                bin, wboxPk + (size_t)i * WPK_U32_PER_LAYER, box_b + i * 256,
                convf, H, W, totalPx);
            fcos_gn_stats<<<64, 256, 0, stream>>>(convf, gmean, grstd, HW);
            fcos_gn_apply<<<elemBlocks, 256, 0, stream>>>(
                convf, gmean, grstd, box_gw + i * 256, box_gb + i * 256,
                bout_, HW, totalPx * 256);
            { unsigned short* t = bin; bin = bout_; bout_ = t; }
        }

        // heads (reuse convf region as the [totalPx][85] logits buffer)
        float* head85 = convf;
        fcos_head_conv<<<(totalPx * 85 + 255) / 256, 256, 0, stream>>>(
            cin, bin, score_w, score_b, pred_w, pred_b, ctr_w, ctr_b,
            head85, H, W);
        fcos_assemble<<<(totalPx * 84 + 255) / 256, 256, 0, stream>>>(
            head85, (float*)d_out, scales, l, H, W, STRD[l], rowBase, totalRows);

        rowBase += HW;
    }
}
